// MultiHeadAttention_9732395892929
// MI455X (gfx1250) — matrix operand order repaired and checked by reading, not compiled, not run on hardware
//
#include <hip/hip_runtime.h>
#include <hip/hip_bf16.h>

// ---------------------------------------------------------------------------
// MHA forward, B=4 S=2048 D=1024 H=16 DH=64, gfx1250 (MI455X), wave32 WMMA.
// All contractions: v_wmma_f32_16x16x32_bf16. Async global->LDS where the
// staging is a pure bf16 copy (ASYNCcnt path), guarded by __has_builtin.
// ---------------------------------------------------------------------------

typedef __attribute__((ext_vector_type(16))) __bf16        bf16x16;
typedef __attribute__((ext_vector_type(16))) unsigned short u16x16;
typedef __attribute__((ext_vector_type(8)))  unsigned int  u32x8;
typedef __attribute__((ext_vector_type(8)))  float         v8f;
typedef __attribute__((ext_vector_type(4)))  int           i32x4;

#define B_   4
#define S_   2048
#define D_   1024
#define H_   16
#define DH_  64

#if defined(__HIP_DEVICE_COMPILE__) && \
    __has_builtin(__builtin_amdgcn_global_load_async_to_lds_b128) && \
    __has_builtin(__builtin_amdgcn_s_wait_asynccnt)
#define ASYNC_LDS 1
#define AS1 __attribute__((address_space(1)))
#define AS3 __attribute__((address_space(3)))
// builtin signature (from clang diagnostic): (AS1 i32x4*, AS3 ..., i32, i32)
#define ASYNC_CP128(gsrc, ldst) \
  __builtin_amdgcn_global_load_async_to_lds_b128( \
      (AS1 i32x4*)(gsrc), (AS3 i32x4*)(ldst), 0, 0)
#else
#define ASYNC_LDS 0
#endif

// fp32 -> bf16 (RNE)
__device__ __forceinline__ unsigned short f32_to_bf16(float f) {
  unsigned int u = __builtin_bit_cast(unsigned int, f);
  u += 0x7FFFu + ((u >> 16) & 1u);
  return (unsigned short)(u >> 16);
}

// two fp32 -> packed bf16x2 dword (v_cvt_pk_bf16_f32 when available)
__device__ __forceinline__ unsigned int pack_bf16x2(float lo, float hi) {
#if defined(__HIP_DEVICE_COMPILE__) && __has_builtin(__builtin_amdgcn_cvt_pk_bf16_f32)
  auto r = __builtin_amdgcn_cvt_pk_bf16_f32(lo, hi);
  return __builtin_bit_cast(unsigned int, r);
#else
  return (unsigned int)f32_to_bf16(lo) | ((unsigned int)f32_to_bf16(hi) << 16);
#endif
}

__device__ __forceinline__ v8f wmma_bf16(bf16x16 a, bf16x16 b, v8f c) {
  return __builtin_amdgcn_wmma_f32_16x16x32_bf16(false, a, false, b,
                                                 (short)0, c, false, false);
}

// ---- Fragment loaders, dword-granular (contiguous k order used for both A and B: k = 16*(lane/16) + s) ----
// A 16x32: lane<16: M=lane, K={0..15}; lane>=16: M=lane-16, K={16..31}
__device__ __forceinline__ bf16x16 lds_load_a32(const unsigned short* sA, int stride,
                                                int m0, int lane) {
  const unsigned int* p =
      (const unsigned int*)(sA + (m0 + (lane & 15)) * stride) + ((lane >> 4) << 3);
  u32x8 u;
#pragma unroll
  for (int i = 0; i < 4; ++i) u[i] = p[i];
#pragma unroll
  for (int i = 0; i < 4; ++i) u[4 + i] = p[4 + i];
  return __builtin_bit_cast(bf16x16, u);
}

// B 32x16 from column-major LDS tile sB[col*stride + k]
__device__ __forceinline__ bf16x16 lds_load_b_colmajor32(const unsigned short* sB,
                                                         int stride, int n0, int lane) {
  const unsigned int* p =
      (const unsigned int*)(sB + (n0 + (lane & 15)) * stride) + ((lane >> 4) << 3);
  u32x8 u;
#pragma unroll
  for (int i = 0; i < 8; ++i) u[i] = p[i];
  return __builtin_bit_cast(bf16x16, u);
}

// B 32x16 from row-major [K][N] LDS tile (non-contiguous: 16-bit gathers)
__device__ __forceinline__ bf16x16 lds_load_b_rowmajor(const unsigned short* sB,
                                                       int stride, int n0, int lane) {
  const unsigned short* p = sB + ((lane >> 4) << 4) * stride + n0 + (lane & 15);
  u16x16 u;
#pragma unroll
  for (int i = 0; i < 16; ++i) u[i] = p[i * stride];
  return __builtin_bit_cast(bf16x16, u);
}

// A fragment from global row-major bf16 [rows][stride]
__device__ __forceinline__ bf16x16 gl_load_a32(const unsigned short* base, int stride,
                                               int row0, int k0, int lane) {
  const unsigned int* p =
      (const unsigned int*)(base + (size_t)(row0 + (lane & 15)) * stride);
  const int kb2 = (k0 >> 1) + ((lane >> 4) << 3);
  u32x8 u;
#pragma unroll
  for (int i = 0; i < 4; ++i) u[i] = p[kb2 + i];
#pragma unroll
  for (int i = 0; i < 4; ++i) u[4 + i] = p[kb2 + 4 + i];
  return __builtin_bit_cast(bf16x16, u);
}

// B fragment for Q@K^T: logical B[d][key], read from K stored [key][DH_]
__device__ __forceinline__ bf16x16 gl_load_bT32(const unsigned short* kmat,
                                                int key0, int d0, int lane) {
  const unsigned int* p = (const unsigned int*)
      (kmat + (size_t)(key0 + (lane & 15)) * DH_ + d0 + ((lane >> 4) << 4));
  u32x8 u;
#pragma unroll
  for (int i = 0; i < 8; ++i) u[i] = p[i];
  return __builtin_bit_cast(bf16x16, u);
}

// ---------------------------------------------------------------------------
// GEMM: 128x128 block tile, K-chunk 32, 8 waves each doing 2x4 WMMA tiles.
// LDS row stride 40 ushorts = 80B: 16B aligned + conflict-free fragment reads.
// ---------------------------------------------------------------------------
#define BM   128
#define BN   128
#define BK   32
#define LDA_ 40
#define LDB_ 40

// out[B,H,S,DH](bf16) = scale * (X(fp32) @ W(fp32) + bias)
__global__ __launch_bounds__(256) void qkv_proj_kernel(
    const float* __restrict__ X, const float* __restrict__ W,
    const float* __restrict__ bias, unsigned short* __restrict__ out,
    float oscale) {
  __shared__ __align__(16) unsigned short sA[BM * LDA_];
  __shared__ __align__(16) unsigned short sB[BN * LDB_];

  const int tid  = threadIdx.x;
  const int lane = tid & 31;
  const int wv   = tid >> 5;
  const int wm   = wv >> 1;
  const int wn   = wv & 1;
  const int m_blk = blockIdx.x * BM;
  const int n_blk = blockIdx.y * BN;

  v8f acc[2][4];
#pragma unroll
  for (int i = 0; i < 2; ++i)
#pragma unroll
    for (int j = 0; j < 4; ++j) acc[i][j] = {};

  for (int kb = 0; kb < D_; kb += BK) {
    // stage A (128x32 fp32 -> bf16), 4 float4/thread, packed dword stores
#pragma unroll
    for (int it = 0; it < 4; ++it) {
      int q  = tid + it * 256;
      int r  = q >> 3;
      int c4 = q & 7;
      const float4 f = *(const float4*)(X + (size_t)(m_blk + r) * D_ + kb + c4 * 4);
      unsigned int* d = (unsigned int*)(sA + r * LDA_ + c4 * 4);
      d[0] = pack_bf16x2(f.x, f.y);
      d[1] = pack_bf16x2(f.z, f.w);
    }
    // stage B (32x128 fp32 -> bf16, column-major scatter)
#pragma unroll
    for (int it = 0; it < 4; ++it) {
      int q  = tid + it * 256;
      int kr = q >> 5;
      int c4 = q & 31;
      const float4 f = *(const float4*)(W + (size_t)(kb + kr) * D_ + n_blk + c4 * 4);
      unsigned int w01 = pack_bf16x2(f.x, f.y);
      unsigned int w23 = pack_bf16x2(f.z, f.w);
      sB[(c4 * 4 + 0) * LDB_ + kr] = (unsigned short)w01;
      sB[(c4 * 4 + 1) * LDB_ + kr] = (unsigned short)(w01 >> 16);
      sB[(c4 * 4 + 2) * LDB_ + kr] = (unsigned short)w23;
      sB[(c4 * 4 + 3) * LDB_ + kr] = (unsigned short)(w23 >> 16);
    }
    if (kb + BK < D_)
      __builtin_prefetch(W + (size_t)(kb + BK + (tid >> 3)) * D_ + n_blk, 0, 1);
    __syncthreads();

    bf16x16 aF[2], bF[4];
#pragma unroll
    for (int mt = 0; mt < 2; ++mt) aF[mt] = lds_load_a32(sA, LDA_, wm * 32 + mt * 16, lane);
#pragma unroll
    for (int nt = 0; nt < 4; ++nt) bF[nt] = lds_load_b_colmajor32(sB, LDB_, wn * 64 + nt * 16, lane);
#pragma unroll
    for (int mt = 0; mt < 2; ++mt)
#pragma unroll
      for (int nt = 0; nt < 4; ++nt)
        acc[mt][nt] = wmma_bf16(aF[mt], bF[nt], acc[mt][nt]);
    __syncthreads();
  }

  const int half = lane >> 4;
  const int ln   = lane & 15;
#pragma unroll
  for (int mt = 0; mt < 2; ++mt) {
#pragma unroll
    for (int nt = 0; nt < 4; ++nt) {
      const int gn = n_blk + wn * 64 + nt * 16 + ln;
      const float bsv = bias[gn];
      const int h = gn >> 6, dh = gn & 63;
#pragma unroll
      for (int j = 0; j < 8; ++j) {
        const int gm = m_blk + wm * 32 + mt * 16 + j + half * 8;
        const int b  = gm >> 11;
        const int s  = gm & (S_ - 1);
        out[(((size_t)(b * H_ + h)) * S_ + s) * DH_ + dh] =
            f32_to_bf16((acc[mt][nt][j] + bsv) * oscale);
      }
    }
  }
}

// out(fp32) = ctx(bf16) @ W(fp32) + bias
__global__ __launch_bounds__(256) void out_proj_kernel(
    const unsigned short* __restrict__ Abf, const float* __restrict__ W,
    const float* __restrict__ bias, float* __restrict__ out) {
  __shared__ __align__(16) unsigned short sA[BM * LDA_];
  __shared__ __align__(16) unsigned short sB[BN * LDB_];

  const int tid  = threadIdx.x;
  const int lane = tid & 31;
  const int wv   = tid >> 5;
  const int wm   = wv >> 1;
  const int wn   = wv & 1;
  const int m_blk = blockIdx.x * BM;
  const int n_blk = blockIdx.y * BN;

  v8f acc[2][4];
#pragma unroll
  for (int i = 0; i < 2; ++i)
#pragma unroll
    for (int j = 0; j < 4; ++j) acc[i][j] = {};

  for (int kb = 0; kb < D_; kb += BK) {
    // stage A: pure bf16 copy -> async global->LDS (b128) when available
#if ASYNC_LDS
#pragma unroll
    for (int it = 0; it < 2; ++it) {
      int q = tid + it * 256;
      int r = q >> 2;                 // 4 x 16B per 32-elem row
      int c = q & 3;
      ASYNC_CP128(Abf + (size_t)(m_blk + r) * D_ + kb + c * 8,
                  sA + r * LDA_ + c * 8);
    }
#else
#pragma unroll
    for (int it = 0; it < 8; ++it) {
      int q  = tid + it * 256;
      int r  = q >> 4;
      int c2 = q & 15;
      *(unsigned int*)(sA + r * LDA_ + c2 * 2) =
          *(const unsigned int*)(Abf + (size_t)(m_blk + r) * D_ + kb + c2 * 2);
    }
#endif
#pragma unroll
    for (int it = 0; it < 4; ++it) {
      int q  = tid + it * 256;
      int kr = q >> 5;
      int c4 = q & 31;
      const float4 f = *(const float4*)(W + (size_t)(kb + kr) * D_ + n_blk + c4 * 4);
      unsigned int w01 = pack_bf16x2(f.x, f.y);
      unsigned int w23 = pack_bf16x2(f.z, f.w);
      sB[(c4 * 4 + 0) * LDB_ + kr] = (unsigned short)w01;
      sB[(c4 * 4 + 1) * LDB_ + kr] = (unsigned short)(w01 >> 16);
      sB[(c4 * 4 + 2) * LDB_ + kr] = (unsigned short)w23;
      sB[(c4 * 4 + 3) * LDB_ + kr] = (unsigned short)(w23 >> 16);
    }
#if ASYNC_LDS
    __builtin_amdgcn_s_wait_asynccnt(0);
#endif
    __syncthreads();

    bf16x16 aF[2], bF[4];
#pragma unroll
    for (int mt = 0; mt < 2; ++mt) aF[mt] = lds_load_a32(sA, LDA_, wm * 32 + mt * 16, lane);
#pragma unroll
    for (int nt = 0; nt < 4; ++nt) bF[nt] = lds_load_b_colmajor32(sB, LDB_, wn * 64 + nt * 16, lane);
#pragma unroll
    for (int mt = 0; mt < 2; ++mt)
#pragma unroll
      for (int nt = 0; nt < 4; ++nt)
        acc[mt][nt] = wmma_bf16(aF[mt], bF[nt], acc[mt][nt]);
    __syncthreads();
  }

  const int half = lane >> 4;
  const int ln   = lane & 15;
#pragma unroll
  for (int mt = 0; mt < 2; ++mt) {
#pragma unroll
    for (int nt = 0; nt < 4; ++nt) {
      const int gn = n_blk + wn * 64 + nt * 16 + ln;
      const float bsv = bias[gn];
#pragma unroll
      for (int j = 0; j < 8; ++j) {
        const int gm = m_blk + wm * 32 + mt * 16 + j + half * 8;
        out[(size_t)gm * D_ + gn] = acc[mt][nt][j] + bsv;
      }
    }
  }
}

// ---------------------------------------------------------------------------
// Flash attention: block = 128 query rows of one (b,h); wave = 16-row strip.
// Q pre-scaled by 1/sqrt(DH) in projection. V staged async to LDS.
// ---------------------------------------------------------------------------
#define VST 72   // 144B rows: 16B-aligned for async b128 writes
#define PST 40

__global__ __launch_bounds__(256) void attention_kernel(
    const unsigned short* __restrict__ Q,   // [B*H, S, DH] bf16 (pre-scaled)
    const unsigned short* __restrict__ K,
    const unsigned short* __restrict__ V,
    unsigned short* __restrict__ ctx) {     // [B, S, H, DH] bf16
  __shared__ __align__(16) unsigned short sV[32 * VST];
  __shared__ __align__(16) unsigned short sP[8 * 16 * PST];

  const int tid   = threadIdx.x;
  const int lane  = tid & 31;
  const int wv    = tid >> 5;
  const int bh    = blockIdx.y;
  const int qbase = blockIdx.x * 128;
  const int qrow0 = qbase + wv * 16;

  const unsigned short* q_ptr = Q + (size_t)bh * S_ * DH_;
  const unsigned short* k_ptr = K + (size_t)bh * S_ * DH_;
  const unsigned short* v_ptr = V + (size_t)bh * S_ * DH_;
  unsigned short* sPw = sP + wv * (16 * PST);

  bf16x16 qF[2];
  qF[0] = gl_load_a32(q_ptr, DH_, qrow0, 0, lane);
  qF[1] = gl_load_a32(q_ptr, DH_, qrow0, 32, lane);

  const int half = lane >> 4;
  const int ln   = lane & 15;

  float m_run[8], l_run[8];
  v8f oacc[4];
#pragma unroll
  for (int j = 0; j < 8; ++j) { m_run[j] = -1e30f; l_run[j] = 0.0f; }
#pragma unroll
  for (int nt = 0; nt < 4; ++nt) oacc[nt] = {};

  const int nchunks = (qbase + 128) >> 5;   // causal: keys <= qbase+127

  for (int c = 0; c < nchunks; ++c) {
    const int key0 = c << 5;

    // stage V chunk (32x64 bf16 = 4KB): one async b128 per thread
#if ASYNC_LDS
    {
      int r  = tid >> 3;                // 8 x 16B per 128B key row
      int cc = tid & 7;
      ASYNC_CP128(v_ptr + (size_t)(key0 + r) * DH_ + cc * 8,
                  sV + r * VST + cc * 8);
    }
#else
#pragma unroll
    for (int it = 0; it < 4; ++it) {
      int q  = tid + it * 256;
      int kr = q >> 5;
      int d2 = q & 31;
      *(unsigned int*)(sV + kr * VST + d2 * 2) =
          *(const unsigned int*)(v_ptr + (size_t)(key0 + kr) * DH_ + d2 * 2);
    }
#endif

    // scores S = Q @ K^T (Q pre-scaled)
    v8f sc[2]; sc[0] = {}; sc[1] = {};
#pragma unroll
    for (int dc = 0; dc < 2; ++dc) {
      bf16x16 bT0 = gl_load_bT32(k_ptr, key0,      dc * 32, lane);
      bf16x16 bT1 = gl_load_bT32(k_ptr, key0 + 16, dc * 32, lane);
      sc[0] = wmma_bf16(qF[dc], bT0, sc[0]);
      sc[1] = wmma_bf16(qF[dc], bT1, sc[1]);
    }

    // causal mask + online softmax (row = j + 8*half spans 16 lanes)
    float pv0[8], pv1[8], mchunk[8];
#pragma unroll
    for (int j = 0; j < 8; ++j) {
      const int qr = qrow0 + j + half * 8;
      float s0 = sc[0][j];
      float s1 = sc[1][j];
      if (key0 + ln > qr)      s0 = -1e30f;
      if (key0 + 16 + ln > qr) s1 = -1e30f;
      pv0[j] = s0; pv1[j] = s1;
      float mx = fmaxf(s0, s1);
#pragma unroll
      for (int d = 1; d < 16; d <<= 1) mx = fmaxf(mx, __shfl_xor(mx, d, 32));
      mchunk[j] = mx;
    }
#pragma unroll
    for (int j = 0; j < 8; ++j) {
      const float mnew  = fmaxf(m_run[j], mchunk[j]);
      const float alpha = __expf(m_run[j] - mnew);
      const float p0 = __expf(pv0[j] - mnew);
      const float p1 = __expf(pv1[j] - mnew);
      float rs = p0 + p1;
#pragma unroll
      for (int d = 1; d < 16; d <<= 1) rs += __shfl_xor(rs, d, 32);
      l_run[j] = l_run[j] * alpha + rs;
      m_run[j] = mnew;
#pragma unroll
      for (int nt = 0; nt < 4; ++nt) oacc[nt][j] *= alpha;
      const int pr = j + half * 8;
      sPw[pr * PST + ln]      = f32_to_bf16(p0);
      sPw[pr * PST + 16 + ln] = f32_to_bf16(p1);
    }
#if ASYNC_LDS
    __builtin_amdgcn_s_wait_asynccnt(0);
#endif
    __syncthreads();    // V staged by all waves; P written

    // O += P @ V
    bf16x16 pF = lds_load_a32(sPw, PST, 0, lane);
#pragma unroll
    for (int nt = 0; nt < 4; ++nt) {
      bf16x16 vF = lds_load_b_rowmajor(sV, VST, nt * 16, lane);
      oacc[nt] = wmma_bf16(pF, vF, oacc[nt]);
    }
    __syncthreads();    // all reads done before next chunk overwrites LDS
  }

  // epilogue: ctx[b, s, h, dh] = O / l
  const int b = bh >> 4;
  const int h = bh & 15;
#pragma unroll
  for (int nt = 0; nt < 4; ++nt) {
    const int dh = nt * 16 + ln;
#pragma unroll
    for (int j = 0; j < 8; ++j) {
      const int s = qrow0 + j + half * 8;
      const float v = oacc[nt][j] / l_run[j];
      ctx[(((size_t)b * S_ + s) * H_ + h) * DH_ + dh] = f32_to_bf16(v);
    }
  }
}

// ---------------------------------------------------------------------------
extern "C" void kernel_launch(void* const* d_in, const int* in_sizes, int n_in,
                              void* d_out, int out_size, void* d_ws, size_t ws_size,
                              hipStream_t stream) {
  (void)in_sizes; (void)n_in; (void)out_size; (void)ws_size;
  const float* queries = (const float*)d_in[0];
  const float* keys    = (const float*)d_in[1];
  const float* values  = (const float*)d_in[2];
  // d_in[3] = causal mask (recomputed analytically in-kernel)
  const float* Wq = (const float*)d_in[4];
  const float* bq = (const float*)d_in[5];
  const float* Wk = (const float*)d_in[6];
  const float* bk = (const float*)d_in[7];
  const float* Wv = (const float*)d_in[8];
  const float* bv = (const float*)d_in[9];
  const float* Wo = (const float*)d_in[10];
  const float* bo = (const float*)d_in[11];
  float* out = (float*)d_out;

  const size_t elems = (size_t)B_ * H_ * S_ * DH_;   // 8M bf16 = 16 MB each
  unsigned short* qb = (unsigned short*)d_ws;
  unsigned short* kb = qb + elems;
  unsigned short* vb = kb + elems;
  unsigned short* cb = vb + elems;

  dim3 blk(256);
  dim3 g1((B_ * S_) / BM, D_ / BN);   // 64 x 8
  qkv_proj_kernel<<<g1, blk, 0, stream>>>(queries, Wq, bq, qb, 0.125f);
  qkv_proj_kernel<<<g1, blk, 0, stream>>>(keys,    Wk, bk, kb, 1.0f);
  qkv_proj_kernel<<<g1, blk, 0, stream>>>(values,  Wv, bv, vb, 1.0f);

  dim3 g2(S_ / 128, B_ * H_);         // 16 x 64
  attention_kernel<<<g2, blk, 0, stream>>>(qb, kb, vb, cb);

  out_proj_kernel<<<g1, blk, 0, stream>>>(cb, Wo, bo, out);
}
